// TransformerModel_29240137351612
// MI455X (gfx1250) — matrix operand order repaired and checked by reading, not compiled, not run on hardware
//
#include <hip/hip_runtime.h>
#include <cstdint>
#include <cstddef>

// ---------------- model constants ----------------
static constexpr int Vv  = 32000;
static constexpr int Dd  = 1024;
static constexpr int NH  = 16;
static constexpr int Ll  = 6;
static constexpr int Tt  = 1024;
static constexpr int Bb  = 2;
static constexpr int HS  = 64;          // Dd / NH
static constexpr int DFF = 4 * Dd;      // 4096
static constexpr int Mrows = Bb * Tt;   // 2048

// ---------------- WMMA types ----------------
typedef __attribute__((ext_vector_type(16))) __bf16 v16bf;
typedef __attribute__((ext_vector_type(8)))  float  v8f;

union AFrag { v16bf v; unsigned int u[8]; };
union BFrag { v16bf v; unsigned int u[8]; };

static __device__ inline v8f wmma_bf16(const v16bf& a, const v16bf& b, const v8f& c) {
    return __builtin_amdgcn_wmma_f32_16x16x32_bf16(
        /*neg_a=*/false, a, /*neg_b=*/false, b,
        /*c_mod=*/(short)0, c, /*reuse_a=*/false, /*reuse_b=*/false);
}

// A-fragment K index for vgpr i, lane-half hh (contiguous order: k = 16*hh + 2*i, matching B)
static __device__ inline int a_kidx(int i, int hh) {
    return 16 * hh + 2 * i;
}

// LDS byte offset of a __shared__ object (for async global->LDS ops)
static __device__ inline unsigned lds_off(const void* p) {
    return (unsigned)(size_t)(__attribute__((address_space(3))) const void*)p;
}

// ---------------- elementwise kernels ----------------
__global__ void cast_f32_bf16_kernel(const float* __restrict__ src,
                                     __bf16* __restrict__ dst, long n) {
    long i = (long)blockIdx.x * blockDim.x + threadIdx.x;
    long stride = (long)gridDim.x * blockDim.x;
    for (; i < n; i += stride) dst[i] = (__bf16)src[i];
}

__global__ void embed_kernel(const int* __restrict__ tokens,
                             const float* __restrict__ tok_emb,
                             const float* __restrict__ pos_emb,
                             float* __restrict__ x) {
    const int bt  = blockIdx.x;          // 0 .. B*T-1
    const int t   = bt % Tt;
    const int tok = tokens[bt];
    for (int d = threadIdx.x; d < Dd; d += blockDim.x)
        x[(size_t)bt * Dd + d] =
            tok_emb[(size_t)tok * Dd + d] + pos_emb[(size_t)t * Dd + d];
}

// LayerNorm over D, fp32 in -> bf16 out (GEMM input precision)
__global__ __launch_bounds__(256)
void ln_kernel(const float* __restrict__ x, const float* __restrict__ sc,
               const float* __restrict__ bi, __bf16* __restrict__ out) {
    const int row = blockIdx.x;
    const int tid = threadIdx.x;
    __shared__ float red[256];
    const float* xr = x + (size_t)row * Dd;

    float acc = 0.f;
    for (int d = tid; d < Dd; d += 256) acc += xr[d];
    red[tid] = acc; __syncthreads();
    for (int off = 128; off > 0; off >>= 1) {
        if (tid < off) red[tid] += red[tid + off];
        __syncthreads();
    }
    const float mean = red[0] * (1.0f / Dd);
    __syncthreads();

    acc = 0.f;
    for (int d = tid; d < Dd; d += 256) { float t = xr[d] - mean; acc += t * t; }
    red[tid] = acc; __syncthreads();
    for (int off = 128; off > 0; off >>= 1) {
        if (tid < off) red[tid] += red[tid + off];
        __syncthreads();
    }
    const float rstd = rsqrtf(red[0] * (1.0f / Dd) + 1e-5f);

    for (int d = tid; d < Dd; d += 256)
        out[(size_t)row * Dd + d] = (__bf16)((xr[d] - mean) * rstd * sc[d] + bi[d]);
}

// ---------------- WMMA GEMM: C = act(A[bf16,MxK] * B[bf16,KxN] + bias) (+res) ----------------
// Block 256 = 8 wave32s; block tile 64x64; each wave a 16x32 output (2 WMMA acc),
// K-step 32. A tile staged via gfx1250 async global->LDS (ASYNCcnt); B tile staged
// transposed via regular loads so fragment K-pairs are contiguous ds_load_b32s.
template <bool HAS_BIAS, bool RELU, bool HAS_RES, bool OUTF, bool OUTB>
__global__ __launch_bounds__(256)
void gemm_bf16_kernel(const __bf16* __restrict__ A, const __bf16* __restrict__ B,
                      const float* __restrict__ bias, const float* res,
                      float* outf, __bf16* outb, int M, int N, int K) {
    const int lane = threadIdx.x & 31;
    const int wave = threadIdx.x >> 5;
    const int wm = wave >> 1;        // 0..3  (16-row slice)
    const int wn = wave & 1;         // 0..1  (32-col slice)
    const int m0 = blockIdx.y * 64;
    const int n0 = blockIdx.x * 64;
    const int mrow = lane & 15;
    const int hh   = lane >> 4;
    const int nn   = lane & 15;

    __shared__ __bf16 As[64][40];    // row-major, padded (async dest)
    __shared__ __bf16 Bs[64][40];    // transposed: [n][k], padded

    v8f acc0, acc1;
#pragma unroll
    for (int r = 0; r < 8; ++r) { acc0[r] = 0.f; acc1[r] = 0.f; }

    const int arow = threadIdx.x >> 2;            // 0..63
    const int aseg = (threadIdx.x & 3) * 8;       // 0,8,16,24
    const int bkk  = threadIdx.x >> 3;            // 0..31
    const int bnc  = (threadIdx.x & 7) * 8;       // 0..56

    for (int k0 = 0; k0 < K; k0 += 32) {
        // stage A tile: async global -> LDS, b128 per lane (ASYNCcnt path)
        {
            const unsigned dst = lds_off(&As[arow][aseg]);
            const __bf16* src = A + (size_t)(m0 + arow) * K + k0 + aseg;
            asm volatile("global_load_async_to_lds_b128 %0, %1, off"
                         :: "v"(dst), "v"(src) : "memory");
        }
        // stage B tile transposed (regular loads + ds stores)
        {
            const __bf16* src = B + (size_t)(k0 + bkk) * N + n0 + bnc;
            __bf16 tmp[8];
            *(uint4*)tmp = *(const uint4*)src;
#pragma unroll
            for (int i = 0; i < 8; ++i) Bs[bnc + i][bkk] = tmp[i];
        }
        if (k0 + 32 < K) {
            __builtin_prefetch(B + (size_t)(k0 + 32 + bkk) * N + n0 + bnc, 0, 1);
        }
        asm volatile("s_wait_asynccnt 0x0" ::: "memory");
        __syncthreads();

        AFrag af; BFrag bf0, bf1;
#pragma unroll
        for (int i = 0; i < 8; ++i)
            af.u[i] = *(const unsigned int*)&As[wm * 16 + mrow][a_kidx(i, hh)];
#pragma unroll
        for (int j = 0; j < 8; ++j) {
            bf0.u[j] = *(const unsigned int*)&Bs[wn * 32 + nn][16 * hh + 2 * j];
            bf1.u[j] = *(const unsigned int*)&Bs[wn * 32 + 16 + nn][16 * hh + 2 * j];
        }

        acc0 = wmma_bf16(af.v, bf0.v, acc0);
        acc1 = wmma_bf16(af.v, bf1.v, acc1);
        __syncthreads();
    }

    // epilogue: two 16x16 tiles per wave
    const int col0 = n0 + wn * 32 + nn;
    const int col1 = col0 + 16;
    float bv0 = 0.f, bv1 = 0.f;
    if (HAS_BIAS) { bv0 = bias[col0]; bv1 = bias[col1]; }
#pragma unroll
    for (int r = 0; r < 8; ++r) {
        const int row = m0 + wm * 16 + r + 8 * hh;
        float v0 = acc0[r] + bv0;
        float v1 = acc1[r] + bv1;
        if (RELU) { v0 = fmaxf(v0, 0.f); v1 = fmaxf(v1, 0.f); }
        if (HAS_RES) {
            v0 += res[(size_t)row * N + col0];
            v1 += res[(size_t)row * N + col1];
        }
        if (OUTF) {
            outf[(size_t)row * N + col0] = v0;
            outf[(size_t)row * N + col1] = v1;
        }
        if (OUTB) {
            outb[(size_t)row * N + col0] = (__bf16)v0;
            outb[(size_t)row * N + col1] = (__bf16)v1;
        }
    }
}

// ---------------- flash attention: one wave per (b, h, 16-query tile) ----------------
__global__ __launch_bounds__(32)
void attn_kernel(const __bf16* __restrict__ qb, const __bf16* __restrict__ kb,
                 const __bf16* __restrict__ vb, __bf16* __restrict__ ob) {
    const int qt = blockIdx.x, h = blockIdx.y, b = blockIdx.z;
    const int q0 = qt * 16;
    const int lane = threadIdx.x;
    const int mrow = lane & 15, hh = lane >> 4, nn = lane & 15;

    __shared__ float  Sld[16][33];
    __shared__ __bf16 Pld[16][36];
    __shared__ float  scl[16];
    __shared__ float  lsm[16];

    // Q fragments (e = 0..31 and 32..63), reused for all key chunks
    AFrag qlo, qhi;
    {
        const __bf16* qrow = qb + ((size_t)(b * Tt + q0 + mrow)) * Dd + h * HS;
#pragma unroll
        for (int i = 0; i < 8; ++i) {
            const int kidx = a_kidx(i, hh);
            qlo.u[i] = *(const unsigned int*)(qrow + kidx);
            qhi.u[i] = *(const unsigned int*)(qrow + 32 + kidx);
        }
    }

    v8f O[4];
#pragma unroll
    for (int j = 0; j < 4; ++j)
#pragma unroll
        for (int r = 0; r < 8; ++r) O[j][r] = 0.f;

    float mprev = -1e30f, lprev = 0.f;   // per-row stats (lanes 0..15 own rows)

    const int nchunk = q0 / 32 + 1;      // causal: keys up to q0+15 covered
    for (int c = 0; c < nchunk; ++c) {
        const int kbase = c * 32;

        // scores S = Q K^T for 32 keys (two 16-col WMMA tiles, K-dim = HS in 2 steps)
        v8f s0, s1;
#pragma unroll
        for (int r = 0; r < 8; ++r) { s0[r] = 0.f; s1[r] = 0.f; }
#pragma unroll
        for (int half = 0; half < 2; ++half) {
            BFrag blo, bhi;
            const __bf16* krow =
                kb + ((size_t)(b * Tt + kbase + half * 16 + nn)) * Dd + h * HS;
#pragma unroll
            for (int j = 0; j < 8; ++j) {
                const int e = 16 * hh + 2 * j;
                blo.u[j] = *(const unsigned int*)(krow + e);
                bhi.u[j] = *(const unsigned int*)(krow + 32 + e);
            }
            if (half == 0) {
                s0 = wmma_bf16(qlo.v, blo.v, s0);
                s0 = wmma_bf16(qhi.v, bhi.v, s0);
            } else {
                s1 = wmma_bf16(qlo.v, blo.v, s1);
                s1 = wmma_bf16(qhi.v, bhi.v, s1);
            }
        }

        // scale + causal mask, spill to LDS
#pragma unroll
        for (int r = 0; r < 8; ++r) {
            const int m = r + 8 * hh, qv = q0 + m;
            float v0 = s0[r] * 0.125f;          // HS^-0.5
            float v1 = s1[r] * 0.125f;
            if (kbase + nn > qv)      v0 = -1e30f;
            if (kbase + 16 + nn > qv) v1 = -1e30f;
            Sld[m][nn]      = v0;
            Sld[m][16 + nn] = v1;
        }
        __syncthreads();

        // online softmax: lanes 0..15 each own one row
        if (lane < 16) {
            float mx = -1e30f;
            for (int i = 0; i < 32; ++i) mx = fmaxf(mx, Sld[lane][i]);
            const float mnew = fmaxf(mprev, mx);
            const float sc   = __expf(mprev - mnew);
            float ssum = 0.f;
            for (int i = 0; i < 32; ++i) {
                const float p = __expf(Sld[lane][i] - mnew);
                ssum += p;
                Pld[lane][i] = (__bf16)p;
            }
            lprev = lprev * sc + ssum;
            mprev = mnew;
            scl[lane] = sc;
        }
        __syncthreads();

        // rescale O accumulators by per-row scale
        float sc8[8];
#pragma unroll
        for (int r = 0; r < 8; ++r) sc8[r] = scl[r + 8 * hh];
#pragma unroll
        for (int j = 0; j < 4; ++j)
#pragma unroll
            for (int r = 0; r < 8; ++r) O[j][r] *= sc8[r];

        // P (16x32) A-fragment from LDS
        AFrag pf;
#pragma unroll
        for (int i = 0; i < 8; ++i)
            pf.u[i] = *(const unsigned int*)&Pld[mrow][a_kidx(i, hh)];

        // O += P * V   (4 HS chunks of 16 columns)
#pragma unroll
        for (int j4 = 0; j4 < 4; ++j4) {
            BFrag vf;
#pragma unroll
            for (int j = 0; j < 8; ++j) {
                const int key = kbase + 16 * hh + 2 * j;
                const __bf16* vp =
                    vb + ((size_t)(b * Tt + key)) * Dd + h * HS + j4 * 16 + nn;
                const unsigned lo = *(const unsigned short*)vp;
                const unsigned hi = *(const unsigned short*)(vp + Dd);
                vf.u[j] = lo | (hi << 16);
            }
            O[j4] = wmma_bf16(pf.v, vf.v, O[j4]);
        }
        __syncthreads();
    }

    if (lane < 16) lsm[lane] = lprev;
    __syncthreads();
    float inv8[8];
#pragma unroll
    for (int r = 0; r < 8; ++r) inv8[r] = 1.0f / lsm[r + 8 * hh];
#pragma unroll
    for (int j4 = 0; j4 < 4; ++j4)
#pragma unroll
        for (int r = 0; r < 8; ++r) {
            const int row = q0 + r + 8 * hh;
            ob[((size_t)(b * Tt + row)) * Dd + h * HS + j4 * 16 + nn] =
                (__bf16)(O[j4][r] * inv8[r]);
        }
}

// ---------------- host orchestration ----------------
extern "C" void kernel_launch(void* const* d_in, const int* in_sizes, int n_in,
                              void* d_out, int out_size, void* d_ws, size_t ws_size,
                              hipStream_t stream) {
    (void)in_sizes; (void)n_in; (void)out_size; (void)ws_size;

    const int*   tokens  = (const int*)  d_in[0];
    const float* tok_emb = (const float*)d_in[1];
    const float* pos_emb = (const float*)d_in[2];
    const float* Wq      = (const float*)d_in[3];
    const float* Wk      = (const float*)d_in[4];
    const float* Wv      = (const float*)d_in[5];
    const float* Wo      = (const float*)d_in[6];
    const float* bo      = (const float*)d_in[7];
    const float* ln1_s   = (const float*)d_in[8];
    const float* ln1_b   = (const float*)d_in[9];
    const float* ln2_s   = (const float*)d_in[10];
    const float* ln2_b   = (const float*)d_in[11];
    const float* W1      = (const float*)d_in[12];
    const float* b1      = (const float*)d_in[13];
    const float* W2      = (const float*)d_in[14];
    const float* b2      = (const float*)d_in[15];
    const float* lnf_s   = (const float*)d_in[16];
    const float* lnf_b   = (const float*)d_in[17];
    const float* Wlm     = (const float*)d_in[18];
    const float* blm     = (const float*)d_in[19];
    float* logits        = (float*)d_out;

    char* ws = (char*)d_ws;
    size_t off = 0;
    auto alloc = [&](size_t bytes) -> void* {
        off = (off + 255) & ~(size_t)255;
        void* p = ws + off;
        off += bytes;
        return p;
    };

    const size_t nDD  = (size_t)Dd * Dd;
    __bf16* wq_bf  = (__bf16*)alloc((size_t)Ll * nDD * 2);
    __bf16* wk_bf  = (__bf16*)alloc((size_t)Ll * nDD * 2);
    __bf16* wv_bf  = (__bf16*)alloc((size_t)Ll * nDD * 2);
    __bf16* wo_bf  = (__bf16*)alloc((size_t)Ll * nDD * 2);
    __bf16* w1_bf  = (__bf16*)alloc((size_t)Ll * Dd * DFF * 2);
    __bf16* w2_bf  = (__bf16*)alloc((size_t)Ll * DFF * Dd * 2);
    __bf16* wlm_bf = (__bf16*)alloc((size_t)Dd * Vv * 2);
    float*  x      = (float*) alloc((size_t)Mrows * Dd * 4);
    __bf16* h_bf   = (__bf16*)alloc((size_t)Mrows * Dd * 2);
    __bf16* q_bf   = (__bf16*)alloc((size_t)Mrows * Dd * 2);
    __bf16* k_bf   = (__bf16*)alloc((size_t)Mrows * Dd * 2);
    __bf16* v_bf   = (__bf16*)alloc((size_t)Mrows * Dd * 2);
    __bf16* at_bf  = (__bf16*)alloc((size_t)Mrows * Dd * 2);
    __bf16* m1_bf  = (__bf16*)alloc((size_t)Mrows * DFF * 2);

    auto cast = [&](const float* s, __bf16* d, long n) {
        cast_f32_bf16_kernel<<<4096, 256, 0, stream>>>(s, d, n);
    };
    cast(Wq,  wq_bf,  (long)Ll * nDD);
    cast(Wk,  wk_bf,  (long)Ll * nDD);
    cast(Wv,  wv_bf,  (long)Ll * nDD);
    cast(Wo,  wo_bf,  (long)Ll * nDD);
    cast(W1,  w1_bf,  (long)Ll * Dd * DFF);
    cast(W2,  w2_bf,  (long)Ll * DFF * Dd);
    cast(Wlm, wlm_bf, (long)Dd * Vv);

    embed_kernel<<<Mrows, 256, 0, stream>>>(tokens, tok_emb, pos_emb, x);

    const dim3 gD(Dd / 64,  Mrows / 64);   // N=1024
    const dim3 gF(DFF / 64, Mrows / 64);   // N=4096
    const dim3 gV(Vv / 64,  Mrows / 64);   // N=32000 -> 500
    const dim3 gAttn(Tt / 16, NH, Bb);

    for (int l = 0; l < Ll; ++l) {
        ln_kernel<<<Mrows, 256, 0, stream>>>(x, ln1_s + (size_t)l * Dd,
                                             ln1_b + (size_t)l * Dd, h_bf);
        gemm_bf16_kernel<false, false, false, false, true><<<gD, 256, 0, stream>>>(
            h_bf, wq_bf + (size_t)l * nDD, nullptr, nullptr, nullptr, q_bf,
            Mrows, Dd, Dd);
        gemm_bf16_kernel<false, false, false, false, true><<<gD, 256, 0, stream>>>(
            h_bf, wk_bf + (size_t)l * nDD, nullptr, nullptr, nullptr, k_bf,
            Mrows, Dd, Dd);
        gemm_bf16_kernel<false, false, false, false, true><<<gD, 256, 0, stream>>>(
            h_bf, wv_bf + (size_t)l * nDD, nullptr, nullptr, nullptr, v_bf,
            Mrows, Dd, Dd);

        attn_kernel<<<gAttn, 32, 0, stream>>>(q_bf, k_bf, v_bf, at_bf);

        // x = x + attn @ Wo + bo
        gemm_bf16_kernel<true, false, true, true, false><<<gD, 256, 0, stream>>>(
            at_bf, wo_bf + (size_t)l * nDD, bo + (size_t)l * Dd, x, x, nullptr,
            Mrows, Dd, Dd);

        ln_kernel<<<Mrows, 256, 0, stream>>>(x, ln2_s + (size_t)l * Dd,
                                             ln2_b + (size_t)l * Dd, h_bf);
        // m1 = relu(h @ W1 + b1)
        gemm_bf16_kernel<true, true, false, false, true><<<gF, 256, 0, stream>>>(
            h_bf, w1_bf + (size_t)l * Dd * DFF, b1 + (size_t)l * DFF,
            nullptr, nullptr, m1_bf, Mrows, DFF, Dd);
        // x = x + m1 @ W2 + b2
        gemm_bf16_kernel<true, false, true, true, false><<<gD, 256, 0, stream>>>(
            m1_bf, w2_bf + (size_t)l * DFF * Dd, b2 + (size_t)l * Dd, x, x, nullptr,
            Mrows, Dd, DFF);
    }

    ln_kernel<<<Mrows, 256, 0, stream>>>(x, lnf_s, lnf_b, h_bf);
    gemm_bf16_kernel<true, false, false, true, false><<<gV, 256, 0, stream>>>(
        h_bf, wlm_bf, blm, nullptr, logits, nullptr, Mrows, Vv, Dd);
}
